// SimOTA_90967407329984
// MI455X (gfx1250) — compile-verified, matrix operand order checked by reading, not run on hardware
//
#include <hip/hip_runtime.h>

// SimOTA detection loss for MI455X (gfx1250, wave32).
// Bandwidth/transcendental-bound workload: cost/IoU matrices are recomputed
// on the fly (never spilled to HBM); top-k selection lives in LDS; the dense
// 80-class QFL reduction is done with v_wmma_f32_16x16x32_f16 row-sums
// (B = ones), 16 anchors per wave, fully-active waves (EXEC==all-1s).
// Hot transcendentals use native v_exp_f32/v_log_f32/v_rcp_f32 instead of
// the IEEE div_scale/div_fixup + guarded-exp sequences (~5x fewer VALU ops
// in the 21.5M-element sigmoid/softplus loops). k2/k3 share the same inlined
// helpers so cost values are bit-identical between threshold selection and
// threshold test.

#define N_IMGC 32
#define T_PERC 32
#define N_CLSC 80
#define A_ANCH 8400            // 80^2 + 40^2 + 20^2
#define D_INC  87              // 3 + 4 + 80
#define NAC    (N_IMGC * A_ANCH)   // 268800 (multiple of 16 and 32)
#define NTC    (N_IMGC * T_PERC)   // 1024
#define BIGC   1000000000.0f
#define CPENC  100000.0f

typedef __attribute__((ext_vector_type(16))) _Float16 v16h;
typedef __attribute__((ext_vector_type(8)))  float    v8f;

// Fast native transcendentals (v_exp_f32 / v_log_f32 / v_rcp_f32).
__device__ __forceinline__ float fexp_(float x) { return __expf(x); }
__device__ __forceinline__ float flog_(float x) { return __logf(x); }
__device__ __forceinline__ float frcp_(float x) { return __builtin_amdgcn_rcpf(x); }

__device__ __forceinline__ float sigmoidf_(float x) {
  return frcp_(1.f + fexp_(-x));
}
// softplus(x) = max(x,0) + log(1 + exp(-|x|))
__device__ __forceinline__ float softplusf_(float x) {
  return fmaxf(x, 0.f) + flog_(1.f + fexp_(-fabsf(x)));
}

// IoU between target box trow[2..5] and predicted box pr[3..6]
__device__ __forceinline__ float iou_ta(const float* pr, const float* trow) {
  float px1 = pr[3], py1 = pr[4], px2 = pr[5], py2 = pr[6];
  float tx1 = trow[2], ty1 = trow[3], tx2 = trow[4], ty2 = trow[5];
  float area_t = (tx2 - tx1) * (ty2 - ty1);
  float area_p = (px2 - px1) * (py2 - py1);
  float iw = fmaxf(fminf(tx2, px2) - fmaxf(tx1, px1), 0.f);
  float ih = fmaxf(fminf(ty2, py2) - fmaxf(ty1, py1), 0.f);
  float inter = iw * ih;
  return inter * frcp_(area_t + area_p - inter);
}

// SimOTA cost(t, a): cls_cost + 3*iou_loss + CENTER_PEN*~in_bc + BIG*~in_box
__device__ __forceinline__ float cost_ta(const float* pr, const float* trow,
                                         float S_a, bool inbox_a, float iou) {
  float stride = pr[2];
  float xc = (pr[0] + 0.5f) * stride;
  float yc = (pr[1] + 0.5f) * stride;
  float tx1 = trow[2], ty1 = trow[3], tx2 = trow[4], ty2 = trow[5];
  bool inb = (xc > tx1) && (yc > ty1) && (xc < tx2) && (yc < ty2);
  float r = 2.5f * stride;
  bool inc = (fabsf(xc - 0.5f * (tx1 + tx2)) < r) &&
             (fabsf(yc - 0.5f * (ty1 + ty2)) < r);
  int cid = (int)trow[1];
  float x = pr[7 + cid];
  // l1p - lp - S = -softplus(x) + softplus(-x) - S
  float cls = softplusf_(-x) - softplusf_(x) - S_a;
  return cls - 3.f * flog_(iou + 1e-8f)
         + ((inb && inc) ? 0.f : CPENC)
         + (inbox_a ? 0.f : BIGC);
}

// K1: per-anchor stats: S = sum log1p(-sigmoid) (f32 accumulation — feeds
// discrete matching), maxp over classes 0..78 (for OOD boost), in_box flag.
__global__ void k1_stats(const float* __restrict__ inp, const float* __restrict__ tgt,
                         float* __restrict__ S, float* __restrict__ maxp,
                         float* __restrict__ inbox) {
  int g = blockIdx.x * blockDim.x + threadIdx.x;
  if (g >= NAC) return;
  int i = g / A_ANCH;
  const float* pr = inp + (size_t)g * D_INC;
  float s = 0.f, lm = 0.f;
  for (int c = 0; c < N_CLSC; ++c) {
    float x = pr[7 + c];
    s -= softplusf_(x);                       // l1p = -softplus(x)
    if (c < N_CLSC - 1) lm = fmaxf(lm, sigmoidf_(x));
  }
  float stride = pr[2];
  float xc = (pr[0] + 0.5f) * stride;
  float yc = (pr[1] + 0.5f) * stride;
  float r = 2.5f * stride;
  bool anyB = false, anyC = false;
  const float* tb = tgt + (size_t)i * T_PERC * 6;
  for (int t = 0; t < T_PERC; ++t) {
    const float* tr = tb + t * 6;
    float tx1 = tr[2], ty1 = tr[3], tx2 = tr[4], ty2 = tr[5];
    anyB = anyB || ((xc > tx1) && (yc > ty1) && (xc < tx2) && (yc < ty2));
    anyC = anyC || ((fabsf(xc - 0.5f * (tx1 + tx2)) < r) &&
                    (fabsf(yc - 0.5f * (ty1 + ty2)) < r));
  }
  S[g] = s;
  maxp[g] = lm;
  inbox[g] = (anyB || anyC) ? 1.f : 0.f;
}

// K2: one block per (image, target) row. Phase A: top-10 masked-IoU sum ->
// dyn_k. Phase B: dyn_k-th smallest cost (stable-sort tie-break by index)
// -> threshold + index. Row cached in LDS (33.6KB of the 320KB/WGP).
__global__ __launch_bounds__(256) void k2_topk(
    const float* __restrict__ inp, const float* __restrict__ tgt,
    const float* __restrict__ S, const float* __restrict__ inbox,
    float* __restrict__ thr, float* __restrict__ kidx, float* __restrict__ dynk) {
  __shared__ float buf[A_ANCH];
  __shared__ float sval[256];
  __shared__ int   sidx[256];
  __shared__ float s_sum, s_thr;
  __shared__ int   s_dk, s_ki;

  int it = blockIdx.x;
  int i = it / T_PERC;
  const float* trow = tgt + (size_t)it * 6;
  int tid = threadIdx.x;

  // Phase A: iou masked by per-anchor in_box
  for (int a = tid; a < A_ANCH; a += 256) {
    const float* pr = inp + ((size_t)i * A_ANCH + a) * D_INC;
    float io = iou_ta(pr, trow);
    buf[a] = (inbox[i * A_ANCH + a] > 0.5f) ? io : 0.f;
  }
  if (tid == 0) s_sum = 0.f;
  __syncthreads();
  for (int k = 0; k < 10; ++k) {
    float bv = -1.f; int bi = 0;
    for (int a = tid; a < A_ANCH; a += 256) {
      float v = buf[a];
      if (v > bv) { bv = v; bi = a; }
    }
    sval[tid] = bv; sidx[tid] = bi;
    __syncthreads();
    for (int s2 = 128; s2 > 0; s2 >>= 1) {
      if (tid < s2) {
        if (sval[tid + s2] > sval[tid]) { sval[tid] = sval[tid + s2]; sidx[tid] = sidx[tid + s2]; }
      }
      __syncthreads();
    }
    if (tid == 0) { s_sum += fmaxf(sval[0], 0.f); buf[sidx[0]] = -1.f; }
    __syncthreads();
  }
  if (tid == 0) {
    int dk = (int)s_sum;                // trunc, like .astype(int32)
    dk = dk < 1 ? 1 : (dk > 10 ? 10 : dk);
    s_dk = dk;
    dynk[it] = (float)dk;
  }
  __syncthreads();
  int dk = s_dk;

  // Phase B: cost row, dk iterations of argmin (ties -> smaller index)
  for (int a = tid; a < A_ANCH; a += 256) {
    const float* pr = inp + ((size_t)i * A_ANCH + a) * D_INC;
    float io = iou_ta(pr, trow);
    bool iba = inbox[i * A_ANCH + a] > 0.5f;
    buf[a] = cost_ta(pr, trow, S[i * A_ANCH + a], iba, io);
  }
  __syncthreads();
  for (int k = 0; k < dk; ++k) {
    float bv = 3.0e38f; int bi = A_ANCH;
    for (int a = tid; a < A_ANCH; a += 256) {
      float v = buf[a];
      if (v < bv || (v == bv && a < bi)) { bv = v; bi = a; }
    }
    sval[tid] = bv; sidx[tid] = bi;
    __syncthreads();
    for (int s2 = 128; s2 > 0; s2 >>= 1) {
      if (tid < s2) {
        if (sval[tid + s2] < sval[tid] ||
            (sval[tid + s2] == sval[tid] && sidx[tid + s2] < sidx[tid])) {
          sval[tid] = sval[tid + s2]; sidx[tid] = sidx[tid + s2];
        }
      }
      __syncthreads();
    }
    if (tid == 0) { s_thr = sval[0]; s_ki = sidx[0]; buf[sidx[0]] = 3.0e38f; }
    __syncthreads();
  }
  if (tid == 0) { thr[it] = s_thr; kidx[it] = (float)s_ki; }
}

// K3: per-anchor matching (threshold test per target + conflict resolution)
__global__ void k3_match(const float* __restrict__ inp, const float* __restrict__ tgt,
                         const float* __restrict__ S, const float* __restrict__ inbox,
                         const float* __restrict__ thr, const float* __restrict__ kidx,
                         float* __restrict__ atype, float* __restrict__ tbox,
                         float* __restrict__ qcls, float* __restrict__ qval) {
  int g = blockIdx.x * blockDim.x + threadIdx.x;
  if (g >= NAC) return;
  int i = g / A_ANCH, a = g % A_ANCH;
  const float* pr = inp + (size_t)g * D_INC;
  float Sa = S[g];
  bool iba = inbox[g] > 0.5f;
  int cnt = 0, firstt = -1, bestt = 0;
  float bestc = 3.4e38f, piou = 0.f;
  for (int t = 0; t < T_PERC; ++t) {
    int it = i * T_PERC + t;
    const float* trow = tgt + (size_t)it * 6;
    float io = iou_ta(pr, trow);
    float c = cost_ta(pr, trow, Sa, iba, io);
    piou = fmaxf(piou, io);                 // p_iou = iou.max over t (unmasked)
    if (c < bestc) { bestc = c; bestt = t; } // argmin, ties -> first t
    float th = thr[it];
    int ki = (int)kidx[it];
    // rank < dyn_k  <=>  cost < thr, or ==thr with index <= k-th pick index
    if (c < th || (c == th && a <= ki)) { cnt++; if (firstt < 0) firstt = t; }
  }
  bool mp = cnt > 0;
  int tp = (cnt > 1) ? bestt : ((cnt == 1) ? firstt : 0);
  const float* trow = tgt + ((size_t)i * T_PERC + tp) * 6;
  int cid = (int)trow[1];
  float at = 0.f;
  if (mp) at = (cid == N_CLSC - 1 && piou > 0.f) ? 2.f : 1.f;  // ood=2, pos=1
  atype[g] = at;
  tbox[4 * (size_t)g + 0] = mp ? trow[2] : 0.f;
  tbox[4 * (size_t)g + 1] = mp ? trow[3] : 0.f;
  tbox[4 * (size_t)g + 2] = mp ? trow[4] : 0.f;
  tbox[4 * (size_t)g + 3] = mp ? trow[5] : 0.f;
  qcls[g] = mp ? (float)cid : -1.f;
  qval[g] = mp ? piou : 0.f;
}

// K4: loss. Each wave owns 16 anchors; QFL over 80 classes is summed with
// v_wmma_f32_16x16x32_f16 (A = per-class QFL values, B = ones -> row sums in
// f32 accumulator). Grid is an exact multiple of 32 threads, no early exits:
// EXEC is all-1s at every WMMA (ISA requirement). CIoU for positives is done
// by lanes 0..15 and reduced with wave32 shuffles.
__global__ __launch_bounds__(256) void k4_loss(
    const float* __restrict__ inp, const float* __restrict__ maxp,
    const float* __restrict__ atype, const float* __restrict__ tbox,
    const float* __restrict__ qcls, const float* __restrict__ qval,
    float* __restrict__ acc) {
  int tid = blockIdx.x * blockDim.x + threadIdx.x;
  int wave = tid >> 5, lane = tid & 31;
  int g = wave * 16 + (lane & 15);          // NAC == 16 * (total waves)
  const float* pr = inp + (size_t)g * D_INC;
  float at = atype[g];
  int qci = (int)qcls[g];
  float qv = qval[g];
  float boost = 2.f * fminf(fmaxf(maxp[g], 0.f), 1.f);
  bool ood = (at == 2.f);

  v16h ones;
  #pragma unroll
  for (int e = 0; e < 16; ++e) ones[e] = (_Float16)1.f;
  v8f accv = {0.f, 0.f, 0.f, 0.f, 0.f, 0.f, 0.f, 0.f};

  // 16-bit A-matrix 16x32 layout: lanes 0-15 hold K={0..7,16..23} of row m,
  // lanes 16-31 hold K={8..15,24..31} of row m (m = lane&15).
  int kofs = (lane & 16) ? 8 : 0;
  #pragma unroll
  for (int cb = 0; cb < 96; cb += 32) {     // classes 0..79, pad 80..95 = 0
    v16h av;
    #pragma unroll
    for (int e = 0; e < 16; ++e) {
      int c = cb + e + ((e < 8) ? 0 : 8) + kofs;
      float val = 0.f;
      if (c < N_CLSC) {
        float x = pr[7 + c];
        float tq = (c == qci && at > 0.f) ? qv : 0.f;
        float p = sigmoidf_(x);
        float d = tq - p;
        float sf = d * d;                              // |t-p|^beta, beta=2
        if (c == N_CLSC - 1 && ood) sf *= boost;       // OOD boost on last cls
        float bce = fmaxf(x, 0.f) - x * tq + flog_(1.f + fexp_(-fabsf(x)));
        val = bce * sf;
      }
      av[e] = (_Float16)val;
    }
    accv = __builtin_amdgcn_wmma_f32_16x16x32_f16(
        false, av, false, ones, (short)0, accv, false, false);
  }
  // D[m][n] = rowsum_m for all n. Lane 0 VGPR r holds rows 0..7 (n=0);
  // lane 16 VGPR r holds rows 8..15 (n=0). Sum those and accumulate.
  float tot = 0.f;
  #pragma unroll
  for (int r2 = 0; r2 < 8; ++r2) tot += accv[r2];
  if (lane == 0 || lane == 16) atomicAdd(&acc[2], tot);

  // CIoU for positives (atype==1 only; ood==2 excluded, matching reference)
  float cv = 0.f, nv = 0.f;
  if (lane < 16 && at == 1.f) {
    float px1 = pr[3], py1 = pr[4], px2 = pr[5], py2 = pr[6];
    const float* tb = &tbox[4 * (size_t)g];
    float tx1 = tb[0], ty1 = tb[1], tx2 = tb[2], ty2 = tb[3];
    float w1 = px2 - px1, h1 = py2 - py1;
    float w2 = tx2 - tx1, h2 = ty2 - ty1;
    float iw = fmaxf(fminf(px2, tx2) - fmaxf(px1, tx1), 0.f);
    float ih = fmaxf(fminf(py2, ty2) - fmaxf(py1, ty1), 0.f);
    float inter = iw * ih;
    float uni = w1 * h1 + w2 * h2 - inter + 1e-7f;
    float iou = inter / uni;
    float cw = fmaxf(px2, tx2) - fminf(px1, tx1);
    float ch = fmaxf(py2, ty2) - fminf(py1, ty1);
    float c2 = cw * cw + ch * ch + 1e-7f;
    float dx = tx1 + tx2 - px1 - px2, dy = ty1 + ty2 - py1 - py2;
    float rho2 = (dx * dx + dy * dy) * 0.25f;
    float da = atanf(w2 / (h2 + 1e-7f)) - atanf(w1 / (h1 + 1e-7f));
    float v = (4.f / (3.14159265358979f * 3.14159265358979f)) * da * da;
    float alpha = v / (v - iou + (1.f + 1e-7f));
    float ciou = iou - (rho2 / c2 + v * alpha);
    cv = 1.f - ciou;
    nv = 1.f;
  }
  #pragma unroll
  for (int o = 16; o > 0; o >>= 1) {
    cv += __shfl_xor(cv, o, 32);
    nv += __shfl_xor(nv, o, 32);
  }
  if (lane == 0) { atomicAdd(&acc[1], cv); atomicAdd(&acc[0], nv); }
}

__global__ void k0_init(float* acc) {
  if (threadIdx.x < 3) acc[threadIdx.x] = 0.f;
}

__global__ void k5_final(const float* __restrict__ acc, float* __restrict__ out,
                         int out_size) {
  float nt = acc[0];
  float denom = fmaxf(nt, 1.f);
  float lbox = 0.2f * acc[1] / denom;
  float lqfl = 0.2f * acc[2] / denom;
  if (out_size > 0) out[0] = lbox + lqfl;
  if (out_size > 1) out[1] = lbox;
  if (out_size > 2) out[2] = lqfl;
}

extern "C" void kernel_launch(void* const* d_in, const int* in_sizes, int n_in,
                              void* d_out, int out_size, void* d_ws, size_t ws_size,
                              hipStream_t stream) {
  (void)in_sizes; (void)n_in; (void)ws_size;
  const float* inp = (const float*)d_in[0];   // (32, 8400, 87) f32
  const float* tgt = (const float*)d_in[1];   // (1024, 6) f32
  float* ws = (float*)d_ws;                   // ~10.8 MB used

  float* S     = ws;
  float* maxp  = ws + (size_t)NAC;
  float* inbox = ws + 2 * (size_t)NAC;
  float* atype = ws + 3 * (size_t)NAC;
  float* tbox  = ws + 4 * (size_t)NAC;        // 4 floats per anchor
  float* qcls  = ws + 8 * (size_t)NAC;
  float* qval  = ws + 9 * (size_t)NAC;
  float* thr   = ws + 10 * (size_t)NAC;
  float* kidx  = thr + NTC;
  float* dynk  = kidx + NTC;
  float* acc   = dynk + NTC;                  // [nt, lbox_sum, lqfl_sum]

  k0_init<<<1, 32, 0, stream>>>(acc);
  k1_stats<<<(NAC + 255) / 256, 256, 0, stream>>>(inp, tgt, S, maxp, inbox);
  k2_topk<<<NTC, 256, 0, stream>>>(inp, tgt, S, inbox, thr, kidx, dynk);
  k3_match<<<(NAC + 255) / 256, 256, 0, stream>>>(inp, tgt, S, inbox, thr, kidx,
                                                  atype, tbox, qcls, qval);
  // 16 anchors per wave -> NAC/16 waves -> *32 threads (exact multiple of 256)
  k4_loss<<<(NAC / 16 * 32) / 256, 256, 0, stream>>>(inp, maxp, atype, tbox,
                                                     qcls, qval, acc);
  k5_final<<<1, 1, 0, stream>>>(acc, (float*)d_out, out_size);
}
